// MPULoss_suggest_1778116461029
// MI455X (gfx1250) — matrix operand order repaired and checked by reading, not compiled, not run on hardware
//
#include <hip/hip_runtime.h>

// ---------------------------------------------------------------------------
// MPULoss on MI455X (gfx1250, wave32, WMMA).
// Heavy op: sims = outputs @ normalize(global_logit)^T  (137 GFLOP) fused with
// per-row max/argmax reduction -> v_wmma_f32_16x16x32_f16.
// B-slab staging now uses GLOBAL_LOAD_ASYNC_TO_LDS_B128 (ASYNCcnt) per
// CDNA5 ISA, waited with s_wait_asynccnt.
// ---------------------------------------------------------------------------

typedef __attribute__((ext_vector_type(16))) _Float16 v16h;
typedef __attribute__((ext_vector_type(8)))  float    v8f;

#define EPS     1e-6f
#define COSEPS  1e-8f
#define NBINS   4096

struct Scal {
  float Psum, nP, nU, sum_pen_U, pu2num, ce_sum, pu4num, selpen;
  float lo, hi;
  int   k_rem, num_sel, ticket, pad;
};

// ---------------------------------------------------------------------------
__device__ __forceinline__ v16h make_v16h(int4 a, int4 b) {
  union { int4 q[2]; v16h h; } u;
  u.q[0] = a; u.q[1] = b;
  return u.h;
}

__device__ __forceinline__ void cvt_store8(_Float16* dst, float4 a, float4 b) {
  union { _Float16 h[8]; int4 q; } u;
  u.h[0] = (_Float16)a.x; u.h[1] = (_Float16)a.y;
  u.h[2] = (_Float16)a.z; u.h[3] = (_Float16)a.w;
  u.h[4] = (_Float16)b.x; u.h[5] = (_Float16)b.y;
  u.h[6] = (_Float16)b.z; u.h[7] = (_Float16)b.w;
  *(int4*)dst = u.q;
}

// CDNA5 async copy: 16B global -> LDS per lane, tracked by ASYNCcnt.
// LDS address = low 32 bits of the flat shared pointer (ISA: addr[31:0]).
__device__ __forceinline__ void async_b128_to_lds(const void* gaddr, void* lds) {
  unsigned lds_off = (unsigned)(size_t)lds;
  unsigned long long ga = (unsigned long long)(size_t)gaddr;
  asm volatile("global_load_async_to_lds_b128 %0, %1, off"
               :: "v"(lds_off), "v"(ga) : "memory");
}

__device__ __forceinline__ void wait_async0() {
  asm volatile("s_wait_asynccnt 0x0" ::: "memory");
}

// ---------------------------------------------------------------------------
// prep: zero scalar accumulators + histogram, Psum = sum(prior), num_sel.
__global__ void prep_kernel(const float* __restrict__ prior,
                            const float* __restrict__ rate,
                            Scal* sc, int* hist, int N, int C) {
  int t = threadIdx.x;
  for (int i = t; i < NBINS; i += 256) hist[i] = 0;
  if (t == 0) {
    sc->nP = 0.f; sc->nU = 0.f; sc->sum_pen_U = 0.f; sc->pu2num = 0.f;
    sc->ce_sum = 0.f; sc->pu4num = 0.f; sc->selpen = 0.f; sc->Psum = 0.f;
    sc->lo = -2.0f; sc->hi = 2.0f;
    int ns = (int)((float)N * rate[0]);     // int(N * 0.1)
    sc->num_sel = ns; sc->k_rem = ns; sc->ticket = 0;
  }
  __syncthreads();
  float ps = 0.f;
  for (int i = t; i < C; i += 256) ps += prior[i];
  atomicAdd(&sc->Psum, ps);
}

__global__ void zero_keys_kernel(unsigned long long* keys, int N) {
  int r = blockIdx.x * blockDim.x + threadIdx.x;
  if (r < N) keys[r] = 0ull;
}

// ---------------------------------------------------------------------------
// Normalize global_logit rows -> f16 (B matrix of the GEMM, L2-resident).
__global__ __launch_bounds__(256)
void gn_kernel(const float* __restrict__ gl, _Float16* __restrict__ gnh, int C) {
  int wid = threadIdx.x >> 5, lane = threadIdx.x & 31;
  int k = blockIdx.x * 8 + wid;
  if (k >= C) return;
  const float* g = gl + (size_t)k * C;
  float x[32]; float ss = 0.f;
  for (int q = 0; q < 32; ++q) { float f = g[q * 32 + lane]; x[q] = f; ss += f * f; }
  for (int off = 1; off < 32; off <<= 1) ss += __shfl_xor(ss, off);
  float inv = 1.0f / fmaxf(sqrtf(ss), COSEPS);
  for (int q = 0; q < 32; ++q)
    gnh[(size_t)k * C + q * 32 + lane] = (_Float16)(x[q] * inv);
}

// ---------------------------------------------------------------------------
// Per-row stats: logZ (softmax log-partition), L2 norm, penalty row sums,
// and the P-row scalar accumulations (pu2 numerator, CE sum, nP/nU).
__global__ __launch_bounds__(256)
void row_stats_kernel(const float* __restrict__ outputs,
                      const int* __restrict__ labels,
                      const float* __restrict__ prior,
                      const int* __restrict__ idxlist,
                      float* __restrict__ logZ, float* __restrict__ denomv,
                      float* __restrict__ rpen, Scal* sc, int N, int C) {
  __shared__ float sIdxw[1024];
  __shared__ float sPrior[1024];
  for (int c = threadIdx.x; c < C; c += 256) {
    sIdxw[c]  = (float)idxlist[c];
    sPrior[c] = prior[c];
  }
  __syncthreads();
  int wid = threadIdx.x >> 5, lane = threadIdx.x & 31;
  int row = blockIdx.x * 8 + wid;
  if (row >= N) return;
  const float* orow = outputs + (size_t)row * C;

  float x[32]; float m = -__builtin_inff(); float ss = 0.f;
  for (int q = 0; q < 32; ++q) {
    float f = orow[q * 32 + lane];
    x[q] = f; m = fmaxf(m, f); ss += f * f;
  }
  for (int off = 1; off < 32; off <<= 1) {
    m  = fmaxf(m, __shfl_xor(m, off));
    ss += __shfl_xor(ss, off);
  }
  float se = 0.f;
  for (int q = 0; q < 32; ++q) se += __expf(x[q] - m);
  for (int off = 1; off < 32; off <<= 1) se += __shfl_xor(se, off);
  float lz = m + __logf(se);

  float pen = 0.f;
  for (int q = 0; q < 32; ++q) {
    int c = q * 32 + lane;
    float soft = __expf(x[q] - lz);
    pen += -__logf(1.0f - soft + EPS) * sIdxw[c];
  }
  for (int off = 1; off < 32; off <<= 1) pen += __shfl_xor(pen, off);

  if (lane == 0) {
    logZ[row]   = lz;
    denomv[row] = fmaxf(sqrtf(ss), COSEPS);
    rpen[row]   = pen;
    int lab = labels[row];
    if (lab < C) {                 // positive row
      float ol   = orow[lab];
      float soft = __expf(ol - lz);
      float lr = -__logf(1.0f - soft + EPS) * sPrior[lab]
                 - __logf(1.0f + EPS) * (sc->Psum - sPrior[lab]);
      atomicAdd(&sc->pu2num, lr);
      atomicAdd(&sc->ce_sum, lz - ol);
      atomicAdd(&sc->nP, 1.0f);
    } else {                       // unlabeled row
      atomicAdd(&sc->nU, 1.0f);
      atomicAdd(&sc->sum_pen_U, pen);
    }
  }
}

// ---------------------------------------------------------------------------
// Fused GEMM + row (max, argmax) reduction.
// Block: 512 threads = 16 waves (4x4), 128x128 output tile, K streamed in
// 32-wide double-buffered LDS slabs. A converted f32->f16 through VGPRs;
// B (already f16) staged via GLOBAL_LOAD_ASYNC_TO_LDS_B128.
#define BM   128
#define BN   128
#define TK   32
#define LDA  40   // padded LDS row stride (halves); keeps 16B alignment

__global__ __launch_bounds__(512)
void sims_gemm_kernel(const float* __restrict__ outputs,
                      const _Float16* __restrict__ gnh,
                      unsigned long long* __restrict__ keys,
                      int N, int C) {
  __shared__ _Float16 As[2][BM * LDA];
  __shared__ _Float16 Bs[2][BN * LDA];

  const int row0 = blockIdx.x * BM;
  const int n0   = blockIdx.y * BN;
  const int t    = threadIdx.x;
  const int wid  = t >> 5, lane = t & 31;
  const int wm   = wid & 3, wn = wid >> 2;      // wave grid 4 (M) x 4 (N)
  const int lm   = lane & 15, lh = lane >> 4;   // lane pos / half
  const int sr   = t >> 2, sq = t & 3;          // staging row / 8-col quad

  v8f acc00 = {}, acc01 = {}, acc10 = {}, acc11 = {};

  // prologue: stage k-slab 0 (A via VGPR convert, B via async-to-LDS)
  {
    const float4* ga = (const float4*)(outputs + (size_t)(row0 + sr) * C + sq * 8);
    async_b128_to_lds(gnh + (size_t)(n0 + sr) * C + sq * 8,
                      &Bs[0][sr * LDA + sq * 8]);
    cvt_store8(&As[0][sr * LDA + sq * 8], ga[0], ga[1]);
    wait_async0();
  }
  __syncthreads();

  const int NK = C / TK;
  for (int kt = 0; kt < NK; ++kt) {
    const int cur = kt & 1, nxt = cur ^ 1;
    const bool havenext = (kt + 1 < NK);
    float4 pa0, pa1;
    if (havenext) {
      const int k0 = (kt + 1) * TK;
      // async B copy for next slab (overlaps with the 4 WMMAs below)
      async_b128_to_lds(gnh + (size_t)(n0 + sr) * C + k0 + sq * 8,
                        &Bs[nxt][sr * LDA + sq * 8]);
      const float4* ga = (const float4*)(outputs + (size_t)(row0 + sr) * C + k0 + sq * 8);
      pa0 = ga[0]; pa1 = ga[1];
      if (kt + 2 < NK)  // pull the slab after next toward L2/L0
        __builtin_prefetch(outputs + (size_t)(row0 + sr) * C + (kt + 2) * TK + sq * 8, 0, 1);
    }

    // ---- compute on current slab ----
    {
      const int aoff = lh * 8;   // A half: K 0..7 / 8..15 (then +16)
      const int boff = lh * 8;   // B half: K 0..7 / 8..15 (then +16)
      v16h a0 = make_v16h(*(const int4*)&As[cur][(wm * 32 + lm) * LDA + aoff],
                          *(const int4*)&As[cur][(wm * 32 + lm) * LDA + 16 + aoff]);
      v16h a1 = make_v16h(*(const int4*)&As[cur][(wm * 32 + 16 + lm) * LDA + aoff],
                          *(const int4*)&As[cur][(wm * 32 + 16 + lm) * LDA + 16 + aoff]);
      v16h b0 = make_v16h(*(const int4*)&Bs[cur][(wn * 32 + lm) * LDA + boff],
                          *(const int4*)&Bs[cur][(wn * 32 + lm) * LDA + boff + 16]);
      v16h b1 = make_v16h(*(const int4*)&Bs[cur][(wn * 32 + 16 + lm) * LDA + boff],
                          *(const int4*)&Bs[cur][(wn * 32 + 16 + lm) * LDA + boff + 16]);
      acc00 = __builtin_amdgcn_wmma_f32_16x16x32_f16(false, a0, false, b0, (short)0, acc00, false, false);
      acc01 = __builtin_amdgcn_wmma_f32_16x16x32_f16(false, a0, false, b1, (short)0, acc01, false, false);
      acc10 = __builtin_amdgcn_wmma_f32_16x16x32_f16(false, a1, false, b0, (short)0, acc10, false, false);
      acc11 = __builtin_amdgcn_wmma_f32_16x16x32_f16(false, a1, false, b1, (short)0, acc11, false, false);
    }

    if (havenext) {
      cvt_store8(&As[nxt][sr * LDA + sq * 8], pa0, pa1);
      wait_async0();   // own async B writes have landed in LDS
    }
    __syncthreads();
  }

  // ---- per-row (max, argmax) over this block's 128-class span ----
  // C/D layout: VGPR p of lanes 0-15 holds (m=p, n=lane), lanes 16-31 (m=p+8).
  for (int i = 0; i < 2; ++i) {
    v8f aj0 = i ? acc10 : acc00;
    v8f aj1 = i ? acc11 : acc01;
    for (int p = 0; p < 8; ++p) {
      float v = aj0[p];
      int   n = n0 + wn * 32 + lm;
      float v1 = aj1[p];
      if (v1 > v) { v = v1; n = n + 16; }           // tie keeps lower index
      for (int off = 1; off < 16; off <<= 1) {      // reduce within 16-lane half
        float ov = __shfl_xor(v, off);
        int   on = __shfl_xor(n, off);
        if (ov > v || (ov == v && on < n)) { v = ov; n = on; }
      }
      if (lm == 0) {
        int row = row0 + wm * 32 + i * 16 + p + lh * 8;
        unsigned int ob = __float_as_uint(v);
        unsigned int ok = (ob & 0x80000000u) ? ~ob : (ob | 0x80000000u);
        unsigned long long key =
            ((unsigned long long)ok << 32) | (unsigned int)(~(unsigned int)n);
        atomicMax(&keys[row], key);                 // cross-block merge
      }
    }
  }
}

// ---------------------------------------------------------------------------
// Unpack merged keys -> per-row max cosine sim (P rows masked to -inf) + idx.
__global__ void unpack_kernel(const unsigned long long* __restrict__ keys,
                              const float* __restrict__ denomv,
                              const int* __restrict__ labels,
                              float* __restrict__ max_sim, int* __restrict__ max_idx,
                              int N, int C) {
  int r = blockIdx.x * blockDim.x + threadIdx.x;
  if (r >= N) return;
  unsigned long long k = keys[r];
  unsigned int kv = (unsigned int)(k >> 32);
  unsigned int fb = (kv & 0x80000000u) ? (kv & 0x7fffffffu) : ~kv;
  float raw = __uint_as_float(fb);
  max_idx[r] = (int)(~(unsigned int)(k & 0xffffffffu));
  float sim = raw / denomv[r];
  max_sim[r] = (labels[r] >= C) ? sim : -__builtin_inff();
}

// ---------------------------------------------------------------------------
// Histogram-based top-k threshold refinement (3 rounds, 4096 bins).
__global__ void hist_kernel(const float* __restrict__ max_sim,
                            const Scal* sc, int* hist, int N) {
  int r = blockIdx.x * blockDim.x + threadIdx.x;
  if (r >= N) return;
  float lo = sc->lo, hi = sc->hi;
  float v = max_sim[r];
  if (v >= lo && v < hi) {
    int b = (int)((v - lo) * ((float)NBINS / (hi - lo)));
    b = min(max(b, 0), NBINS - 1);
    atomicAdd(&hist[b], 1);
  }
}

__global__ void scan_kernel(Scal* sc, int* hist) {
  if (threadIdx.x != 0 || blockIdx.x != 0) return;
  int krem = sc->k_rem;
  float lo = sc->lo, hi = sc->hi;
  float w = (hi - lo) * (1.0f / (float)NBINS);
  int cum = 0, b = 0;
  for (b = NBINS - 1; b > 0; --b) {
    int c = hist[b];
    if (cum + c >= krem) break;
    cum += c;
  }
  sc->k_rem = krem - cum;
  sc->lo = lo + w * (float)b;
  sc->hi = lo + w * (float)(b + 1);
  for (int i = 0; i < NBINS; ++i) hist[i] = 0;
}

// ---------------------------------------------------------------------------
// Apply selection: accumulate pu4 numerator and selected-row penalty sum.
__global__ void accum_kernel(const float* __restrict__ max_sim,
                             const int* __restrict__ max_idx,
                             const float* __restrict__ logZ,
                             const float* __restrict__ rpen,
                             const float* __restrict__ outputs,
                             Scal* sc, int N, int C) {
  int r = blockIdx.x * blockDim.x + threadIdx.x;
  if (r >= N) return;
  float v = max_sim[r];               // -inf for P rows
  float lo = sc->lo, hi = sc->hi;
  bool sel = false;
  if (v >= hi) sel = true;
  else if (v >= lo) sel = (atomicAdd(&sc->ticket, 1) < sc->k_rem);
  if (sel) {
    atomicAdd(&sc->pu4num, logZ[r] - outputs[(size_t)r * C + max_idx[r]]);
    atomicAdd(&sc->selpen, rpen[r]);
  }
}

__global__ void finalize_kernel(const Scal* sc, float* out, int L) {
  if (threadIdx.x != 0 || blockIdx.x != 0) return;
  float nP = sc->nP, nU = sc->nU;
  float ns = (float)sc->num_sel;
  float pu3 = (sc->sum_pen_U - sc->selpen) / fmaxf(1.0f, nU - ns) / (float)L;
  float pu2 = -sc->pu2num / fmaxf(1.0f, nP);
  float PUL = pu3 + pu2;              // pu1 statically skipped: C == L
  float pulw = PUL * 1.0f;            // PU_W
  float pu4 = sc->pu4num / ns;
  float cross = sc->ce_sum / nP;      // 0/0 -> nan when nP == 0 (matches ref)
  float obj = __builtin_isnan(cross) ? (2.0f * pulw + pu4)
                                     : (2.0f * pulw + 2.0f * cross + pu4);
  out[0] = obj; out[1] = pulw; out[2] = cross; out[3] = pu4;
}

// ---------------------------------------------------------------------------
extern "C" void kernel_launch(void* const* d_in, const int* in_sizes, int n_in,
                              void* d_out, int out_size, void* d_ws, size_t ws_size,
                              hipStream_t stream) {
  const float* outputs = (const float*)d_in[0];
  const int*   labels  = (const int*)d_in[1];
  const float* prior   = (const float*)d_in[2];
  const int*   idxlist = (const int*)d_in[3];
  const float* gl      = (const float*)d_in[4];
  const float* rate    = (const float*)d_in[5];

  const int C = in_sizes[2];          // 1024
  const int N = in_sizes[0] / C;      // 65536
  const int L = in_sizes[3];          // 1024
  float* out = (float*)d_out;

  size_t off = 0;
  auto alloc = [&](size_t bytes) -> char* {
    char* p = (char*)d_ws + off;
    off = (off + bytes + 255) & ~(size_t)255;
    return p;
  };
  unsigned long long* keys = (unsigned long long*)alloc((size_t)N * 8);
  float* max_sim = (float*)alloc((size_t)N * 4);
  int*   max_idx = (int*)alloc((size_t)N * 4);
  float* denomv  = (float*)alloc((size_t)N * 4);
  float* logZ    = (float*)alloc((size_t)N * 4);
  float* rpen    = (float*)alloc((size_t)N * 4);
  int*   hist    = (int*)alloc(NBINS * 4);
  Scal*  sc      = (Scal*)alloc(sizeof(Scal));
  _Float16* gnh  = (_Float16*)alloc((size_t)C * C * 2);
  (void)ws_size; (void)n_in; (void)out_size;

  prep_kernel<<<1, 256, 0, stream>>>(prior, rate, sc, hist, N, C);
  zero_keys_kernel<<<(N + 255) / 256, 256, 0, stream>>>(keys, N);
  gn_kernel<<<C / 8, 256, 0, stream>>>(gl, gnh, C);
  row_stats_kernel<<<N / 8, 256, 0, stream>>>(outputs, labels, prior, idxlist,
                                              logZ, denomv, rpen, sc, N, C);
  dim3 g(N / BM, C / BN);
  sims_gemm_kernel<<<g, 512, 0, stream>>>(outputs, gnh, keys, N, C);
  unpack_kernel<<<(N + 255) / 256, 256, 0, stream>>>(keys, denomv, labels,
                                                     max_sim, max_idx, N, C);
  for (int it = 0; it < 3; ++it) {
    hist_kernel<<<(N + 255) / 256, 256, 0, stream>>>(max_sim, sc, hist, N);
    scan_kernel<<<1, 1, 0, stream>>>(sc, hist);
  }
  accum_kernel<<<(N + 255) / 256, 256, 0, stream>>>(max_sim, max_idx, logZ, rpen,
                                                    outputs, sc, N, C);
  finalize_kernel<<<1, 1, 0, stream>>>(sc, out, L);
}
